// ConditioningEncoder_61014305407432
// MI455X (gfx1250) — compile-verified, matrix operand order checked by reading, not run on hardware
//
#include <hip/hip_runtime.h>
#include <hip/hip_bf16.h>

#define HIDDEN 256
#define CONDC  512
#define NHEADS 4
#define HD     64
#define BATCH  16
#define TT     2048
#define TS     512

typedef __bf16 bf16_t;
typedef __bf16 v16bf __attribute__((ext_vector_type(16)));
typedef float  v8f   __attribute__((ext_vector_type(8)));
typedef unsigned int v4u __attribute__((ext_vector_type(4)));
typedef int v4i __attribute__((ext_vector_type(4)));
typedef int v8i __attribute__((ext_vector_type(8)));

#if __has_builtin(__builtin_amdgcn_tensor_load_to_lds)
#define HAVE_TDM 1
#else
#define HAVE_TDM 0
#endif

// Native conversions: backend picks v_cvt_*_bf16 forms (far fewer VALU than
// a hand-rolled round-to-nearest-even sequence).
static __device__ __forceinline__ bf16_t f2bf(float f) { return (bf16_t)f; }
static __device__ __forceinline__ float  bf2f(bf16_t b) { return (float)b; }

static __device__ __forceinline__ v8f vzero8() {
  v8f v;
#pragma unroll
  for (int i = 0; i < 8; ++i) v[i] = 0.0f;
  return v;
}

static __device__ __forceinline__ v16bf ld16(const bf16_t* p) {
  return *reinterpret_cast<const v16bf*>(p);   // callers guarantee 32B alignment
}

#if HAVE_TDM
// 2D TDM descriptor (ISA 8.3/8.4): copy `rows` rows of `tile_w` bf16 elements,
// global row stride `stride_e` elements, packed into LDS at byte offset ldsOff.
static __device__ __forceinline__ void tdm_load_2d_bf16(
    unsigned ldsOff, const void* gsrc, unsigned tile_w, unsigned rows, unsigned stride_e) {
  unsigned long long ga = (unsigned long long)(size_t)gsrc;
  v4u g0;
  g0[0] = 1u;                                                  // count=1, user D#
  g0[1] = ldsOff;                                              // lds_addr (bytes)
  g0[2] = (unsigned)ga;                                        // global_addr[31:0]
  g0[3] = (unsigned)((ga >> 32) & 0x01FFFFFFu) | (2u << 30);   // addr[56:32] | type=2
  v8i g1;
  g1[0] = (int)(1u << 16);                                     // data_size=1 (2B), mask=0
  g1[1] = (int)((tile_w & 0xFFFFu) << 16);                     // tensor_dim0[15:0] @63:48
  g1[2] = (int)((tile_w >> 16) | ((rows & 0xFFFFu) << 16));    // dim0 hi | dim1 lo
  g1[3] = (int)((rows >> 16) | ((tile_w & 0xFFFFu) << 16));    // dim1 hi | tile_dim0
  g1[4] = (int)(rows & 0xFFFFu);                               // tile_dim1 | tile_dim2=0
  g1[5] = (int)stride_e;                                       // tensor_dim0_stride lo32
  g1[6] = 0;
  g1[7] = 0;
  v4i gz;
#pragma unroll
  for (int i = 0; i < 4; ++i) gz[i] = 0;
#if __clang_major__ >= 23
  v8i gz8;
#pragma unroll
  for (int i = 0; i < 8; ++i) gz8[i] = 0;
  __builtin_amdgcn_tensor_load_to_lds(g0, g1, gz, gz, gz8, 0);
#else
  __builtin_amdgcn_tensor_load_to_lds(g0, g1, gz, gz, 0);
#endif
}
#endif

// ---------------------------------------------------------------------------
// Convert + transpose:  in f32 [B, K, N]  ->  out bf16 [B, N, K]
// ---------------------------------------------------------------------------
__global__ void cvt_transpose_kernel(const float* __restrict__ in,
                                     bf16_t* __restrict__ out, int K, int N) {
  __shared__ float tile[32][33];
  const int b  = blockIdx.z;
  const int n0 = blockIdx.x * 32;
  const int k0 = blockIdx.y * 32;
  const int tx = threadIdx.x, ty = threadIdx.y;
  const float* ib = in + (size_t)b * K * N;
  bf16_t*      ob = out + (size_t)b * N * K;
#pragma unroll
  for (int i = 0; i < 4; ++i)
    tile[ty + i * 8][tx] = ib[(size_t)(k0 + ty + i * 8) * N + n0 + tx];
  __syncthreads();
#pragma unroll
  for (int i = 0; i < 4; ++i)
    ob[(size_t)(n0 + ty + i * 8) * K + k0 + tx] = f2bf(tile[tx][ty + i * 8]);
}

__global__ void cvt_w_kernel(const float* __restrict__ in, bf16_t* __restrict__ out, int n) {
  int i = blockIdx.x * blockDim.x + threadIdx.x;
  if (i < n) out[i] = f2bf(in[i]);
}

__global__ void fuse_w_kernel(bf16_t* __restrict__ Outw, const float* __restrict__ A,
                              const float* __restrict__ Bm, int M, int K, int N) {
  int idx = blockIdx.x * blockDim.x + threadIdx.x;
  if (idx >= M * N) return;
  int m = idx / N, n = idx % N;
  float acc = 0.0f;
  for (int k = 0; k < K; ++k) acc += A[(size_t)m * K + k] * Bm[(size_t)k * N + n];
  Outw[idx] = f2bf(acc);
}

__global__ void fuse_b_kernel(float* __restrict__ Outb, const float* __restrict__ A,
                              const float* __restrict__ bin, const float* __restrict__ badd,
                              int M, int K) {
  int m = blockIdx.x * blockDim.x + threadIdx.x;
  if (m >= M) return;
  float acc = badd[m];
  for (int k = 0; k < K; ++k) acc += A[(size_t)m * K + k] * bin[k];
  Outb[m] = acc;
}

// ---------------------------------------------------------------------------
// Transposed-output GEMM:  YT[b][tok][cout] = XT[b][tok][:] @ W[cout][:] + bias
// ---------------------------------------------------------------------------
__global__ __launch_bounds__(128) void gemm_tn_kernel(
    const bf16_t* __restrict__ XT, const bf16_t* __restrict__ W,
    const float* __restrict__ bias, bf16_t* __restrict__ YT,
    int Ntok, int K, int Cout) {
  const int lane = threadIdx.x & 31;
  const int wave = threadIdx.x >> 5;
  const int hl   = lane >> 4;
  const int l16  = lane & 15;
  const int b    = blockIdx.z;
  const int m0   = blockIdx.y * 128 + wave * 32;
  const int n0   = blockIdx.x * 64;
  const bf16_t* Xb = XT + (size_t)b * Ntok * K;

  v8f acc[2][4];
#pragma unroll
  for (int mt = 0; mt < 2; ++mt)
#pragma unroll
    for (int nc = 0; nc < 4; ++nc) acc[mt][nc] = vzero8();

  for (int k0 = 0; k0 < K; k0 += 32) {
    v16bf a[2], bm[4];
#pragma unroll
    for (int mt = 0; mt < 2; ++mt) {
      const bf16_t* p = Xb + (size_t)(m0 + mt * 16 + l16) * K + k0 + hl * 16;
      a[mt] = ld16(p);
      if (k0 + 32 < K) __builtin_prefetch(p + 32, 0, 1);
    }
#pragma unroll
    for (int nc = 0; nc < 4; ++nc)
      bm[nc] = ld16(W + (size_t)(n0 + nc * 16 + l16) * K + k0 + hl * 16);
#pragma unroll
    for (int mt = 0; mt < 2; ++mt)
#pragma unroll
      for (int nc = 0; nc < 4; ++nc)
        acc[mt][nc] = __builtin_amdgcn_wmma_f32_16x16x32_bf16(
            false, a[mt], false, bm[nc], (short)0, acc[mt][nc], false, false);
  }

  bf16_t* Yb = YT + (size_t)b * Ntok * Cout;
#pragma unroll
  for (int mt = 0; mt < 2; ++mt)
#pragma unroll
    for (int nc = 0; nc < 4; ++nc) {
      int col = n0 + nc * 16 + l16;
      float bs = bias[col];
#pragma unroll
      for (int r = 0; r < 8; ++r) {
        int tok = m0 + mt * 16 + hl * 8 + r;
        Yb[(size_t)tok * Cout + col] = f2bf(acc[mt][nc][r] + bs);  // coalesced
      }
    }
}

// ---------------------------------------------------------------------------
// Normal-output GEMM:  Y[b][m][n] = W[m][:] @ XT[b][n][:] + bias[m]   (bf16)
// ---------------------------------------------------------------------------
__global__ __launch_bounds__(128) void gemm_nn_kernel(
    const bf16_t* __restrict__ W, const bf16_t* __restrict__ XT,
    const float* __restrict__ bias, bf16_t* __restrict__ Y,
    int M, int K, int N) {
  const int lane = threadIdx.x & 31;
  const int wave = threadIdx.x >> 5;
  const int hl   = lane >> 4;
  const int l16  = lane & 15;
  const int b    = blockIdx.z;
  const int m0   = blockIdx.y * 128 + wave * 32;
  const int n0   = blockIdx.x * 64;
  const bf16_t* Xb = XT + (size_t)b * N * K;

  v8f acc[2][4];
#pragma unroll
  for (int mt = 0; mt < 2; ++mt)
#pragma unroll
    for (int nc = 0; nc < 4; ++nc) acc[mt][nc] = vzero8();

  for (int k0 = 0; k0 < K; k0 += 32) {
    v16bf a[2], bm[4];
#pragma unroll
    for (int mt = 0; mt < 2; ++mt)
      a[mt] = ld16(W + (size_t)(m0 + mt * 16 + l16) * K + k0 + hl * 16);
#pragma unroll
    for (int nc = 0; nc < 4; ++nc) {
      const bf16_t* p = Xb + (size_t)(n0 + nc * 16 + l16) * K + k0 + hl * 16;
      bm[nc] = ld16(p);
      if (k0 + 32 < K) __builtin_prefetch(p + 32, 0, 1);
    }
#pragma unroll
    for (int mt = 0; mt < 2; ++mt)
#pragma unroll
      for (int nc = 0; nc < 4; ++nc)
        acc[mt][nc] = __builtin_amdgcn_wmma_f32_16x16x32_bf16(
            false, a[mt], false, bm[nc], (short)0, acc[mt][nc], false, false);
  }

  bf16_t* Yb = Y + (size_t)b * M * N;
#pragma unroll
  for (int mt = 0; mt < 2; ++mt)
#pragma unroll
    for (int r = 0; r < 8; ++r) {
      int m = m0 + mt * 16 + hl * 8 + r;
      float bs = bias[m];
#pragma unroll
      for (int nc = 0; nc < 4; ++nc) {
        int n = n0 + nc * 16 + l16;
        Yb[(size_t)m * N + n] = f2bf(acc[mt][nc][r] + bs);       // coalesced
      }
    }
}

// ---------------------------------------------------------------------------
// In-place RoPE on transposed bf16 tensor [B, Ntok, 256].
// ---------------------------------------------------------------------------
__global__ void rope_kernel(bf16_t* __restrict__ Q, int Ntok) {
  size_t idx   = (size_t)blockIdx.x * blockDim.x + threadIdx.x;
  size_t total = (size_t)BATCH * Ntok * NHEADS * 32;
  if (idx >= total) return;
  int j  = (int)(idx & 31);
  size_t r = idx >> 5;
  int hh = (int)(r % NHEADS); r /= NHEADS;
  int t  = (int)(r % Ntok);
  int b  = (int)(r / Ntok);

  float inv = __powf(10000.0f, -(float)j / 32.0f);
  float sn, cs;
  __sincosf((float)t * inv, &sn, &cs);

  size_t i1 = ((size_t)b * Ntok + t) * HIDDEN + hh * HD + j;
  size_t i2 = i1 + 32;
  float x1 = bf2f(Q[i1]), x2 = bf2f(Q[i2]);
  Q[i1] = f2bf(x1 * cs - x2 * sn);
  Q[i2] = f2bf(x2 * cs + x1 * sn);
}

// ---------------------------------------------------------------------------
// Flash cross-attention with TDM-staged K tiles.
//   qT [B,Tt,256], kT [B,Ts,256], v [B,256,Ts] -> oT [B,Tt,256]
// Block = 4 waves, each wave owns 16 query rows; the K head tile is staged
// into LDS in two 256-row chunks by the Tensor Data Mover and reused by all
// 4 waves (4x L2-traffic reduction + latency hidden behind TENSORcnt).
// Grid (TT/64, NHEADS, B), block 128.  LDS: 32KB K-stage + 4KB P.
// ---------------------------------------------------------------------------
__global__ __launch_bounds__(128) void attn_kernel(
    const bf16_t* __restrict__ QT, const bf16_t* __restrict__ KT,
    const bf16_t* __restrict__ Vn, const float* __restrict__ cmask,
    bf16_t* __restrict__ OT) {
  __shared__ bf16_t LdsK[256 * HD];      // 32 KB: 256 s-rows x 64 channels
  __shared__ bf16_t Pl[4][16 * 32];      // 4 KB: per-wave P staging
  const int lane = threadIdx.x & 31;
  const int wave = threadIdx.x >> 5;
  const int hl   = lane >> 4;
  const int l16  = lane & 15;
  const int b    = blockIdx.z;
  const int hh   = blockIdx.y;
  const int t0   = (blockIdx.x * 4 + wave) * 16;

  const bf16_t* khead = KT + (size_t)b * TS * HIDDEN + hh * HD;  // row stride HIDDEN

  // Q A-frags: row t = t0+l16, channels contiguous
  v16bf aq[2];
  {
    const bf16_t* qrow = QT + ((size_t)b * TT + t0 + l16) * HIDDEN + hh * HD;
#pragma unroll
    for (int j = 0; j < 2; ++j) aq[j] = ld16(qrow + j * 32 + hl * 16);
  }

  v8f of[4];
#pragma unroll
  for (int i = 0; i < 4; ++i) of[i] = vzero8();
  float rm[8], rl[8];
#pragma unroll
  for (int r = 0; r < 8; ++r) { rm[r] = -3.0e38f; rl[r] = 0.0f; }

  for (int sb = 0; sb < 2; ++sb) {       // two 256-row K stages
    // ---- stage K rows [sb*256, sb*256+256) into LDS ----
#if HAVE_TDM
    if (wave == 0) {
      tdm_load_2d_bf16((unsigned)(size_t)&LdsK[0],
                       khead + (size_t)sb * 256 * HIDDEN,
                       HD, 256, HIDDEN);
      __builtin_amdgcn_s_wait_tensorcnt(0);
    }
#else
    for (int i = threadIdx.x; i < 256 * HD / 16; i += 128) {
      int row = i >> 2, part = i & 3;
      *reinterpret_cast<v16bf*>(&LdsK[row * HD + part * 16]) =
          ld16(khead + (size_t)(sb * 256 + row) * HIDDEN + part * 16);
    }
#endif
    __syncthreads();

    for (int s0l = 0; s0l < 256; s0l += 32) {
      const int s0 = sb * 256 + s0l;

      // ---- S chunk (16x32) as two 16x16 WMMA tiles, K from LDS ----
      v8f sf[2];
#pragma unroll
      for (int sc = 0; sc < 2; ++sc) {
        v8f accs = vzero8();
        const bf16_t* krow = &LdsK[(s0l + sc * 16 + l16) * HD];
#pragma unroll
        for (int j = 0; j < 2; ++j) {
          v16bf bk = *reinterpret_cast<const v16bf*>(krow + j * 32 + hl * 16);
          accs = __builtin_amdgcn_wmma_f32_16x16x32_bf16(
              false, aq[j], false, bk, (short)0, accs, false, false);
        }
        sf[sc] = accs;
      }

      const float cm0 = cmask[(size_t)b * TS + s0 + l16];
      const float cm1 = cmask[(size_t)b * TS + s0 + 16 + l16];
#pragma unroll
      for (int r = 0; r < 8; ++r) {
        float v0 = sf[0][r] * 0.125f; if (cm0 == 0.0f) v0 = -1.0e4f;
        float v1 = sf[1][r] * 0.125f; if (cm1 == 0.0f) v1 = -1.0e4f;
        sf[0][r] = v0; sf[1][r] = v1;
      }

      // ---- online softmax ----
      float nm[8], alpha[8];
#pragma unroll
      for (int r = 0; r < 8; ++r) {
        float mx = fmaxf(sf[0][r], sf[1][r]);
#pragma unroll
        for (int d = 1; d < 16; d <<= 1) mx = fmaxf(mx, __shfl_xor(mx, d, 32));
        nm[r]    = fmaxf(rm[r], mx);
        alpha[r] = __expf(rm[r] - nm[r]);
        rm[r]    = nm[r];
      }
#pragma unroll
      for (int r = 0; r < 8; ++r) {
        float p0 = __expf(sf[0][r] - nm[r]);
        float p1 = __expf(sf[1][r] - nm[r]);
        sf[0][r] = p0; sf[1][r] = p1;
        float s = p0 + p1;
#pragma unroll
        for (int d = 1; d < 16; d <<= 1) s += __shfl_xor(s, d, 32);
        rl[r] = rl[r] * alpha[r] + s;
      }
#pragma unroll
      for (int f = 0; f < 4; ++f)
#pragma unroll
        for (int r = 0; r < 8; ++r) of[f][r] *= alpha[r];

      // ---- P: C-layout -> LDS -> A-layout (per-wave region, block-sync'd) ----
      __syncthreads();
#pragma unroll
      for (int sc = 0; sc < 2; ++sc)
#pragma unroll
        for (int r = 0; r < 8; ++r)
          Pl[wave][(hl * 8 + r) * 32 + sc * 16 + l16] = f2bf(sf[sc][r]);
      __syncthreads();
      v16bf ap = ld16(&Pl[wave][l16 * 32 + hl * 16]);

      // ---- O += P @ V^T : V frags contiguous from global ----
#pragma unroll
      for (int ct = 0; ct < 4; ++ct) {
        const bf16_t* vp = Vn + ((size_t)b * HIDDEN + hh * HD + ct * 16 + l16) * TS + s0 + hl * 16;
        v16bf bv = ld16(vp);
        of[ct] = __builtin_amdgcn_wmma_f32_16x16x32_bf16(
            false, ap, false, bv, (short)0, of[ct], false, false);
      }
    }
    __syncthreads();   // protect LdsK before next stage overwrites it
  }

  // finalize -> oT[t][c] (coalesced bf16)
#pragma unroll
  for (int r = 0; r < 8; ++r) {
    int t = t0 + hl * 8 + r;
    bf16_t* orow = OT + ((size_t)b * TT + t) * HIDDEN + hh * HD;
    float inv = 1.0f / rl[r];
#pragma unroll
    for (int ct = 0; ct < 4; ++ct)
      orow[ct * 16 + l16] = f2bf(of[ct][r] * inv);
  }
}

// ---------------------------------------------------------------------------
// Fused output projection + FiLM.
// ---------------------------------------------------------------------------
__global__ __launch_bounds__(128) void film_gemm_kernel(
    const bf16_t* __restrict__ Gw, const bf16_t* __restrict__ Bw,
    const float* __restrict__ gb, const float* __restrict__ bb,
    const bf16_t* __restrict__ OT, const float* __restrict__ Xin,
    const float* __restrict__ xmask, float* __restrict__ Out) {
  const int lane = threadIdx.x & 31;
  const int wave = threadIdx.x >> 5;
  const int hl   = lane >> 4;
  const int l16  = lane & 15;
  const int b    = blockIdx.z;
  const int m0   = blockIdx.y * 64 + wave * 16;
  const int n0   = blockIdx.x * 64;
  const bf16_t* Ob = OT + (size_t)b * TT * HIDDEN;

  v8f ag[4], abv[4];
#pragma unroll
  for (int i = 0; i < 4; ++i) { ag[i] = vzero8(); abv[i] = vzero8(); }

  for (int k0 = 0; k0 < HIDDEN; k0 += 32) {
    v16bf wg = ld16(Gw + (size_t)(m0 + l16) * HIDDEN + k0 + hl * 16);
    v16bf wb = ld16(Bw + (size_t)(m0 + l16) * HIDDEN + k0 + hl * 16);
#pragma unroll
    for (int nc = 0; nc < 4; ++nc) {
      const bf16_t* p = Ob + (size_t)(n0 + nc * 16 + l16) * HIDDEN + k0 + hl * 16;
      v16bf xm = ld16(p);
      if (k0 + 32 < HIDDEN) __builtin_prefetch(p + 32, 0, 1);
      ag[nc]  = __builtin_amdgcn_wmma_f32_16x16x32_bf16(
          false, wg, false, xm, (short)0, ag[nc], false, false);
      abv[nc] = __builtin_amdgcn_wmma_f32_16x16x32_bf16(
          false, wb, false, xm, (short)0, abv[nc], false, false);
    }
  }

#pragma unroll
  for (int r = 0; r < 8; ++r) {
    int m = m0 + hl * 8 + r;
    float gB0 = gb[m], bB0 = bb[m];
    const float* xrow = Xin + ((size_t)b * HIDDEN + m) * TT;
    float*       orow = Out + ((size_t)b * HIDDEN + m) * TT;
#pragma unroll
    for (int nc = 0; nc < 4; ++nc) {
      int t = n0 + nc * 16 + l16;
      float g  = ag[nc][r]  + gB0;
      float be = abv[nc][r] + bB0;
      orow[t] = (xrow[t] * g + be) * xmask[(size_t)b * TT + t];   // coalesced
    }
  }
}

// ---------------------------------------------------------------------------
extern "C" void kernel_launch(void* const* d_in, const int* in_sizes, int n_in,
                              void* d_out, int out_size, void* d_ws, size_t ws_size,
                              hipStream_t stream) {
  const float* x      = (const float*)d_in[0];
  const float* x_mask = (const float*)d_in[1];
  const float* cond   = (const float*)d_in[2];
  const float* cmask  = (const float*)d_in[3];
  const float* w_cond = (const float*)d_in[4];
  const float* b_cond = (const float*)d_in[5];
  const float* wq     = (const float*)d_in[6];
  const float* bq     = (const float*)d_in[7];
  const float* wk     = (const float*)d_in[8];
  const float* bk     = (const float*)d_in[9];
  const float* wv     = (const float*)d_in[10];
  const float* bv     = (const float*)d_in[11];
  const float* wo     = (const float*)d_in[12];
  const float* bo     = (const float*)d_in[13];
  const float* wf     = (const float*)d_in[14];
  const float* bfi    = (const float*)d_in[15];

  bf16_t* xT     = (bf16_t*)d_ws;
  bf16_t* condT  = xT     + (size_t)BATCH * TT * HIDDEN;
  bf16_t* qT     = condT  + (size_t)BATCH * TS * CONDC;
  bf16_t* kT     = qT     + (size_t)BATCH * TT * HIDDEN;
  bf16_t* vb     = kT     + (size_t)BATCH * TS * HIDDEN;
  bf16_t* oT     = vb     + (size_t)BATCH * HIDDEN * TS;
  bf16_t* wq_bf  = oT     + (size_t)BATCH * TT * HIDDEN;
  bf16_t* wkf_bf = wq_bf  + (size_t)HIDDEN * HIDDEN;
  bf16_t* wvf_bf = wkf_bf + (size_t)HIDDEN * CONDC;
  bf16_t* gW_bf  = wvf_bf + (size_t)HIDDEN * CONDC;
  bf16_t* bW_bf  = gW_bf  + (size_t)HIDDEN * HIDDEN;
  float*  bkf    = (float*)(bW_bf + (size_t)HIDDEN * HIDDEN);
  float*  bvf    = bkf + HIDDEN;
  float*  gB     = bvf + HIDDEN;
  float*  bB     = gB  + HIDDEN;

  // 1) layout conversion + weight fusion (tiny, one-shot)
  cvt_transpose_kernel<<<dim3(TT / 32, HIDDEN / 32, BATCH), dim3(32, 8), 0, stream>>>(x,    xT,    HIDDEN, TT);
  cvt_transpose_kernel<<<dim3(TS / 32, CONDC  / 32, BATCH), dim3(32, 8), 0, stream>>>(cond, condT, CONDC,  TS);
  cvt_w_kernel<<<(HIDDEN * HIDDEN + 255) / 256, 256, 0, stream>>>(wq, wq_bf, HIDDEN * HIDDEN);
  fuse_w_kernel<<<(HIDDEN * CONDC + 255) / 256, 256, 0, stream>>>(wkf_bf, wk, w_cond, HIDDEN, HIDDEN, CONDC);
  fuse_w_kernel<<<(HIDDEN * CONDC + 255) / 256, 256, 0, stream>>>(wvf_bf, wv, w_cond, HIDDEN, HIDDEN, CONDC);
  fuse_w_kernel<<<(HIDDEN * HIDDEN + 255) / 256, 256, 0, stream>>>(gW_bf, wf, wo, HIDDEN, HIDDEN, HIDDEN);
  fuse_w_kernel<<<(HIDDEN * HIDDEN + 255) / 256, 256, 0, stream>>>(bW_bf, wf + (size_t)HIDDEN * HIDDEN, wo, HIDDEN, HIDDEN, HIDDEN);
  fuse_b_kernel<<<1, 256, 0, stream>>>(bkf, wk, b_cond, bk, HIDDEN, HIDDEN);
  fuse_b_kernel<<<1, 256, 0, stream>>>(bvf, wv, b_cond, bv, HIDDEN, HIDDEN);
  fuse_b_kernel<<<1, 256, 0, stream>>>(gB, wf, bo, bfi, HIDDEN, HIDDEN);
  fuse_b_kernel<<<1, 256, 0, stream>>>(bB, wf + (size_t)HIDDEN * HIDDEN, bo, bfi + HIDDEN, HIDDEN, HIDDEN);

  // 2) projections
  dim3 blk(128);
  gemm_tn_kernel<<<dim3(HIDDEN / 64, TT / 128, BATCH), blk, 0, stream>>>(xT,    wq_bf,  bq,  qT, TT, HIDDEN, HIDDEN);
  gemm_tn_kernel<<<dim3(HIDDEN / 64, TS / 128, BATCH), blk, 0, stream>>>(condT, wkf_bf, bkf, kT, TS, CONDC,  HIDDEN);
  gemm_nn_kernel<<<dim3(TS / 64, HIDDEN / 128, BATCH), blk, 0, stream>>>(wvf_bf, condT, bvf, vb, HIDDEN, CONDC, TS);

  // 3) RoPE (in-place, pairwise, bf16)
  {
    size_t n1 = (size_t)BATCH * TT * NHEADS * 32;
    rope_kernel<<<(unsigned)((n1 + 255) / 256), 256, 0, stream>>>(qT, TT);
    size_t n2 = (size_t)BATCH * TS * NHEADS * 32;
    rope_kernel<<<(unsigned)((n2 + 255) / 256), 256, 0, stream>>>(kT, TS);
  }

  // 4) flash cross-attention (TDM-staged K)
  attn_kernel<<<dim3(TT / 64, NHEADS, BATCH), blk, 0, stream>>>(qT, kT, vb, cmask, oT);

  // 5) fused output projection + FiLM
  film_gemm_kernel<<<dim3(TT / 64, HIDDEN / 64, BATCH), blk, 0, stream>>>(
      gW_bf, bW_bf, gB, bB, oT, x, x_mask, (float*)d_out);
}
